// L2gating_47966194762112
// MI455X (gfx1250) — matrix operand order repaired and checked by reading, not compiled, not run on hardware
//
#include <hip/hip_runtime.h>

// Problem constants (match the reference)
#define N_PTS 16384
#define DIM   2048
#define NG    4096

// GEMM tiling: 256x256 block tile, 512 threads (16 waves), each wave 64x64.
#define BM 256
#define BN 256
#define BK 32
#define KTILES (DIM / BK)   // 64
#define THREADS 512

typedef __attribute__((ext_vector_type(16))) __bf16 bf16x16;
typedef __attribute__((ext_vector_type(8)))  float  f32x8;
typedef __attribute__((ext_vector_type(4)))  unsigned int u32x4;

// Pack two fp32 -> two bf16 (round-half-up) in a single v_perm_b32:
// result = { hi16(b + 0x8000), hi16(a + 0x8000) }  (low half = a).
// perm treats {src0:src1} as 8 bytes (src1 = bytes 0-3); sel 0x07060302 picks
// bytes {a.b2, a.b3, b.b2, b.b3}.
__device__ __forceinline__ unsigned pkbf(float a, float b) {
    unsigned ua = __float_as_uint(a) + 0x8000u;
    unsigned ub = __float_as_uint(b) + 0x8000u;
    return __builtin_amdgcn_perm(ub, ua, 0x07060302u);
}

// LDS byte offset of 16B chunk q (0..3) of a 64B row, XOR-swizzled so that
// 16-lane b128 fragment loads are bank-conflict-free.
__device__ __forceinline__ int swz(int row, int q) {
    return row * 64 + (((q ^ (row & 3) ^ ((row >> 2) & 3)) & 3) << 4);
}

// Per-thread staging registers for one 256x32 fp32 tile (2 chunks x 32B).
struct Stage { float4 v[2][2]; };

__device__ __forceinline__ void ldg_tile(const float* __restrict__ base, int k0,
                                         Stage& s, int tid) {
#pragma unroll
    for (int i = 0; i < 2; ++i) {
        int g   = tid + i * THREADS;     // chunk id 0..1023
        int row = g >> 2;                // 0..255
        int q   = g & 3;                 // 16B chunk within 64B row
        const float4* p =
            reinterpret_cast<const float4*>(base + (size_t)row * DIM + k0 + q * 8);
        s.v[i][0] = p[0];
        s.v[i][1] = p[1];
    }
}

// Convert staged fp32 -> packed bf16 and store to LDS (conversion happens here,
// after the WMMA block, so prefetched loads aren't waited on early).
__device__ __forceinline__ void sts_tile(unsigned char* lds, const Stage& s, int tid) {
#pragma unroll
    for (int i = 0; i < 2; ++i) {
        int g   = tid + i * THREADS;
        int row = g >> 2;
        int q   = g & 3;
        u32x4 u;
        u[0] = pkbf(s.v[i][0].x, s.v[i][0].y);
        u[1] = pkbf(s.v[i][0].z, s.v[i][0].w);
        u[2] = pkbf(s.v[i][1].x, s.v[i][1].y);
        u[3] = pkbf(s.v[i][1].z, s.v[i][1].w);
        *reinterpret_cast<u32x4*>(lds + swz(row, q)) = u;
    }
}

// Load one 16x32 (A) / 32x16 (B) bf16 fragment. q0/q1 select which 16B K-chunks
// this half-wave owns per the CDNA5 VGPR layout tables.
__device__ __forceinline__ bf16x16 ld_frag(const unsigned char* lds, int row,
                                           int q0, int q1) {
    union { bf16x16 v; u32x4 u[2]; } f;
    f.u[0] = *reinterpret_cast<const u32x4*>(lds + swz(row, q0));
    f.u[1] = *reinterpret_cast<const u32x4*>(lds + swz(row, q1));
    return f.v;
}

// ---------------------------------------------------------------------------
// Row sum-of-squares: one 256-thread block per row.
// ---------------------------------------------------------------------------
__global__ void __launch_bounds__(256)
rownorm_kernel(const float* __restrict__ src, float* __restrict__ dst, int cols) {
    const int row = blockIdx.x;
    const float4* p = reinterpret_cast<const float4*>(src + (size_t)row * cols);
    float s = 0.0f;
    for (int i = threadIdx.x; i < (cols >> 2); i += 256) {
        float4 v = p[i];
        s = fmaf(v.x, v.x, s);
        s = fmaf(v.y, v.y, s);
        s = fmaf(v.z, v.z, s);
        s = fmaf(v.w, v.w, s);
    }
    __shared__ float red[256];
    red[threadIdx.x] = s;
    __syncthreads();
    for (int off = 128; off > 0; off >>= 1) {
        if (threadIdx.x < off) red[threadIdx.x] += red[threadIdx.x + off];
        __syncthreads();
    }
    if (threadIdx.x == 0) dst[row] = red[0];
}

// ---------------------------------------------------------------------------
// Fused GEMM + distance epilogue.
// out[r][c] = -sqrt(max(x2[r] + e2[c] - 2 * dot(x[r], emb[c]), 0))
// ---------------------------------------------------------------------------
__global__ void __launch_bounds__(THREADS, 1)
l2_gemm_kernel(const float* __restrict__ x, const float* __restrict__ emb,
               const float* __restrict__ x2, const float* __restrict__ e2,
               float* __restrict__ out) {
    __shared__ __align__(16) unsigned char lA[2][BM * 64];  // 2 x 16 KB
    __shared__ __align__(16) unsigned char lB[2][BN * 64];  // 2 x 16 KB

    const int tid   = threadIdx.x;
    const int lane  = tid & 31;
    const int wave  = tid >> 5;      // 0..15
    const int waveM = wave >> 2;     // 0..3  -> 64-row slab
    const int waveN = wave & 3;      // 0..3  -> 64-col slab
    const int kh    = lane >> 4;     // half-wave id (K-chunk selector)
    const int l15   = lane & 15;

    const float* abase = x   + (size_t)blockIdx.y * BM * DIM;
    const float* bbase = emb + (size_t)blockIdx.x * BN * DIM;

    Stage sA, sB;
    ldg_tile(abase, 0, sA, tid);
    ldg_tile(bbase, 0, sB, tid);
    sts_tile(lA[0], sA, tid);
    sts_tile(lB[0], sB, tid);
    __syncthreads();

    f32x8 acc[4][4] = {};  // 16 x v8f accumulators = 64x64 per wave

#pragma unroll 1
    for (int kt = 0; kt < KTILES; ++kt) {
        const int buf = kt & 1;

        // Prefetch next K-chunk from global into registers (overlaps WMMA).
        if (kt + 1 < KTILES) {
            ldg_tile(abase, (kt + 1) * BK, sA, tid);
            ldg_tile(bbase, (kt + 1) * BK, sB, tid);
        }

        bf16x16 afr, bfr[4];
#pragma unroll
        for (int j = 0; j < 4; ++j)   // B 32x16 frag: chunks (kh, 2+kh)
            bfr[j] = ld_frag(lB[buf], waveN * 64 + j * 16, kh, 2 + kh);

#pragma unroll
        for (int i = 0; i < 4; ++i) {
            // A 16x32 frag: chunks (kh, 2+kh); streamed one tile at a time.
            afr = ld_frag(lA[buf], waveM * 64 + i * 16, kh, 2 + kh);
#pragma unroll
            for (int j = 0; j < 4; ++j)
                acc[i][j] = __builtin_amdgcn_wmma_f32_16x16x32_bf16(
                    false, afr, false, bfr[j], (short)0, acc[i][j],
                    false, false);
        }

        if (kt + 1 < KTILES) {
            // Safe: all waves finished reading buf^1 before the previous barrier.
            sts_tile(lA[buf ^ 1], sA, tid);
            sts_tile(lB[buf ^ 1], sB, tid);
            __syncthreads();
        }
    }

    // Epilogue. C/D layout: lane<16 -> (M=j, N=lane); lane>=16 -> (M=j+8, N=lane-16).
    const int rowBase = blockIdx.y * BM + waveM * 64;
    const int colBase = blockIdx.x * BN + waveN * 64;
#pragma unroll
    for (int i = 0; i < 4; ++i) {
        const int r0 = rowBase + i * 16 + kh * 8;
        float xs[8];
#pragma unroll
        for (int j = 0; j < 8; ++j) xs[j] = x2[r0 + j];
#pragma unroll
        for (int jn = 0; jn < 4; ++jn) {
            const int c = colBase + jn * 16 + l15;
            const float ee = e2[c];
#pragma unroll
            for (int j = 0; j < 8; ++j) {
                float sq = fmaf(-2.0f, acc[i][jn][j], xs[j] + ee);
                sq = fmaxf(sq, 0.0f);
                out[(size_t)(r0 + j) * NG + c] = -sqrtf(sq);
            }
        }
    }
}

// ---------------------------------------------------------------------------
extern "C" void kernel_launch(void* const* d_in, const int* in_sizes, int n_in,
                              void* d_out, int out_size, void* d_ws, size_t ws_size,
                              hipStream_t stream) {
    (void)in_sizes; (void)n_in; (void)out_size; (void)ws_size;
    const float* x   = (const float*)d_in[0];
    const float* emb = (const float*)d_in[1];
    float* out = (float*)d_out;
    float* ws  = (float*)d_ws;
    float* x2  = ws;            // N_PTS floats
    float* e2  = ws + N_PTS;    // NG floats   (total 80 KB of d_ws)

    rownorm_kernel<<<N_PTS, 256, 0, stream>>>(x,   x2, DIM);
    rownorm_kernel<<<NG,    256, 0, stream>>>(emb, e2, DIM);

    dim3 grid(NG / BN, N_PTS / BM);  // (16, 64)
    l2_gemm_kernel<<<grid, THREADS, 0, stream>>>(x, emb, x2, e2, out);
}
